// Net_37443524887171
// MI455X (gfx1250) — matrix operand order repaired and checked by reading, not compiled, not run on hardware
//
#include <hip/hip_runtime.h>
#include <hip/hip_bf16.h>
#include <cstdint>
#include <cstddef>

#define HIDDEN 1024
#define SSIZE  6
#define L1_OUT 1030          // HIDDEN + SSIZE
#define NB     8192
#define NA     32
#define L2_IN  2048

typedef __attribute__((ext_vector_type(16))) _Float16 v16h;
typedef __attribute__((ext_vector_type(8)))  _Float16 v8h;
typedef __attribute__((ext_vector_type(4)))  _Float16 v4h;
typedef __attribute__((ext_vector_type(8)))  float    v8f;

// ---------------------------------------------------------------------------
// Phase 1: feature transform. One block (256 thr, 8 wave32) per batch row.
// Each thread owns 4 of the 1024 hidden columns; the 6 PSQT columns collapse
// to the single column 1024+s (b1 cancels in x2-y2). All W1 reads are
// coalesced float2 (rows are 8B aligned: 1030*4 = 4120 bytes).
// ---------------------------------------------------------------------------
__global__ __launch_bounds__(256) void ft_kernel(
    const int*   __restrict__ xi, const int* __restrict__ yi,
    const float* __restrict__ v,  const int* __restrict__ s,
    const float* __restrict__ W1, const float* __restrict__ b1,
    _Float16* __restrict__ act, float* __restrict__ psqt)
{
  __shared__ int   sx[NA];
  __shared__ int   sy[NA];
  __shared__ float sv[NA];
  __shared__ int   ssel;

  const int b = blockIdx.x;
  const int t = threadIdx.x;

  if (t < NA)          { sx[t] = xi[b * NA + t]; sv[t] = v[b * NA + t]; }
  else if (t < 2 * NA) { sy[t - NA] = yi[b * NA + (t - NA)]; }
  else if (t == 2 * NA){ ssel = s[b]; }
  __syncthreads();

  const int col = 4 * t;
  float4 accx = *reinterpret_cast<const float4*>(b1 + col);  // b1 + 16*t : 16B aligned
  float4 accy = accx;

  #pragma unroll 8
  for (int a = 0; a < NA; ++a) {
    const float va = sv[a];
    const float* rx = W1 + (size_t)sx[a] * L1_OUT + col;
    const float* ry = W1 + (size_t)sy[a] * L1_OUT + col;
    float2 x0 = *reinterpret_cast<const float2*>(rx);
    float2 x1 = *reinterpret_cast<const float2*>(rx + 2);
    float2 y0 = *reinterpret_cast<const float2*>(ry);
    float2 y1 = *reinterpret_cast<const float2*>(ry + 2);
    accx.x += va * x0.x; accx.y += va * x0.y;
    accx.z += va * x1.x; accx.w += va * x1.y;
    accy.x += va * y0.x; accy.y += va * y0.y;
    accy.z += va * y1.x; accy.w += va * y1.y;
  }

  // PSQT term: only column 1024+sel is ever selected; b1 cancels.
  if (t < NA) {
    const int sel = ssel;
    float pp = sv[t] * (W1[(size_t)sx[t] * L1_OUT + HIDDEN + sel]
                      - W1[(size_t)sy[t] * L1_OUT + HIDDEN + sel]);
    #pragma unroll
    for (int off = 16; off > 0; off >>= 1) pp += __shfl_xor(pp, off, 32);
    if (t == 0) psqt[b] = pp * 0.08f;   // 0.5 / PSQT_COEFF
  }

  v4h hx = { (_Float16)fmaxf(accx.x, 0.f), (_Float16)fmaxf(accx.y, 0.f),
             (_Float16)fmaxf(accx.z, 0.f), (_Float16)fmaxf(accx.w, 0.f) };
  v4h hy = { (_Float16)fmaxf(accy.x, 0.f), (_Float16)fmaxf(accy.y, 0.f),
             (_Float16)fmaxf(accy.z, 0.f), (_Float16)fmaxf(accy.w, 0.f) };
  *reinterpret_cast<v4h*>(act + (size_t)b * L2_IN + col)          = hx;
  *reinterpret_cast<v4h*>(act + (size_t)b * L2_IN + HIDDEN + col) = hy;
}

// ---------------------------------------------------------------------------
// W2 (6x2048 f32) -> f16, zero-padded to 16x2048 (WMMA N tile).
// ---------------------------------------------------------------------------
__global__ __launch_bounds__(256) void prep_w2(const float* __restrict__ W2,
                                               _Float16* __restrict__ w2h)
{
  const int i = blockIdx.x * 256 + threadIdx.x;   // 16*2048 = 32768 total
  const int n = i >> 11;
  const int k = i & 2047;
  const float val = (n < SSIZE) ? W2[n * L2_IN + k] : 0.f;
  w2h[i] = (_Float16)val;
}

// ---------------------------------------------------------------------------
// Phase 2: head GEMM via v_wmma_f32_16x16x32_f16.
// Block = 16 batch rows; 8 waves split K=2048 (8 WMMA each), partial C tiles
// reduced through LDS; epilogue selects column s[b], adds bias+PSQT, sigmoid.
// Fragment layouts per CDNA5 ISA 7.12.2:
//   A (16x32 f16): lane<16 -> M=lane, K 0..7 (v0-3) & 16..23 (v4-7);
//                  lane>=16 -> K 8..15 & 24..31.
//   B (32x16 f16): lane<16 -> N=lane, K 0..7 & 16..23; lane>=16 -> K 8..15 & 24..31.
// ---------------------------------------------------------------------------
__global__ __launch_bounds__(256) void head_kernel(
    const _Float16* __restrict__ act, const _Float16* __restrict__ w2h,
    const float* __restrict__ psqt, const int* __restrict__ s,
    const float* __restrict__ b2, float* __restrict__ out)
{
  __shared__ float cred[8 * 256];
  __shared__ float ctile[16 * 16];

  const int b0   = blockIdx.x * 16;
  const int t    = threadIdx.x;
  const int w    = t >> 5;
  const int lane = t & 31;
  const int half = lane >> 4;     // lane half selects K sub-range
  const int mn   = lane & 15;     // A row / B column

  const int k0 = w * 256;
  const _Float16* arow = act + (size_t)(b0 + mn) * L2_IN + k0;
  const _Float16* brow = w2h + (size_t)mn * L2_IN + k0;

  v8f c = {};
  #pragma unroll
  for (int it = 0; it < 8; ++it) {
    const int kk = it * 32;

    const _Float16* pa = arow + kk + half * 8;
    v8h alo = *reinterpret_cast<const v8h*>(pa);        // K base+0..7   (16B aligned)
    v8h ahi = *reinterpret_cast<const v8h*>(pa + 16);   // K base+16..23
    v16h afrag = __builtin_shufflevector(alo, ahi,
        0,1,2,3,4,5,6,7,8,9,10,11,12,13,14,15);

    const _Float16* pb = brow + kk + half * 8;
    v8h blo = *reinterpret_cast<const v8h*>(pb);        // K base+0..7
    v8h bhi = *reinterpret_cast<const v8h*>(pb + 16);   // K base+16..23
    v16h bfrag = __builtin_shufflevector(blo, bhi,
        0,1,2,3,4,5,6,7,8,9,10,11,12,13,14,15);

    c = __builtin_amdgcn_wmma_f32_16x16x32_f16(
        /*neg_a=*/false, afrag, /*neg_b=*/false, bfrag,
        /*c_mod=*/(short)0, c, /*reuse_a=*/false, /*reuse_b=*/false);
  }

  #pragma unroll
  for (int j = 0; j < 8; ++j) cred[w * 256 + lane * 8 + j] = c[j];
  __syncthreads();

  // Sum 8 wave partials; map (VGPR j, lane) -> (M,N) of the 16x16 C tile.
  {
    const int l2 = t >> 3;        // lane position 0..31
    const int j  = t & 7;         // accumulator VGPR index
    float sum = 0.f;
    #pragma unroll
    for (int ww = 0; ww < 8; ++ww) sum += cred[ww * 256 + l2 * 8 + j];
    const int m = j + ((l2 >= 16) ? 8 : 0);
    const int n = l2 & 15;
    ctile[m * 16 + n] = sum;
  }
  __syncthreads();

  if (t < 16) {
    const int m   = t;
    const int sel = s[b0 + m];
    const float val = ctile[m * 16 + sel] + b2[sel] + psqt[b0 + m];
    out[b0 + m] = 1.f / (1.f + __expf(-val));
  }
}

// ---------------------------------------------------------------------------
extern "C" void kernel_launch(void* const* d_in, const int* in_sizes, int n_in,
                              void* d_out, int out_size, void* d_ws, size_t ws_size,
                              hipStream_t stream)
{
  const int*   x  = (const int*)  d_in[0];
  const int*   y  = (const int*)  d_in[1];
  const float* v  = (const float*)d_in[2];
  const int*   s  = (const int*)  d_in[3];
  const float* W1 = (const float*)d_in[4];
  const float* b1 = (const float*)d_in[5];
  const float* W2 = (const float*)d_in[6];
  const float* b2 = (const float*)d_in[7];
  float* out = (float*)d_out;

  char* ws = (char*)d_ws;
  _Float16* act  = (_Float16*)ws;                                   // 32 MiB
  _Float16* w2h  = (_Float16*)(ws + (size_t)NB * L2_IN * 2);        // 64 KiB
  float*    psqt = (float*)   (ws + (size_t)NB * L2_IN * 2
                                  + (size_t)16 * L2_IN * 2);        // 32 KiB

  prep_w2   <<<dim3((16 * L2_IN) / 256), dim3(256), 0, stream>>>(W2, w2h);
  ft_kernel <<<dim3(NB),                 dim3(256), 0, stream>>>(x, y, v, s, W1, b1, act, psqt);
  head_kernel<<<dim3(NB / 16),           dim3(256), 0, stream>>>(act, w2h, psqt, s, b2, out);
}
